// LeNet_8624294331041
// MI455X (gfx1250) — matrix operand order repaired and checked by reading, not compiled, not run on hardware
//
#include <hip/hip_runtime.h>
#include <hip/hip_bf16.h>

typedef __attribute__((ext_vector_type(16))) _Float16 v16h;
typedef __attribute__((ext_vector_type(8)))  _Float16 v8h;
typedef __attribute__((ext_vector_type(8)))  float    v8f;

// ---------------------------------------------------------------------------
// Sizes (fixed by the reference)
// ---------------------------------------------------------------------------
#define N5 524288
#define N4 65536
#define N3 8192
#define N2 1024          // after 3rd pool: N3/8
#define KNB 27

// ---------------------------------------------------------------------------
// init: zero BN stats region of workspace
// ---------------------------------------------------------------------------
__global__ void zero_kernel(float* p, int n) {
  int i = blockIdx.x * blockDim.x + threadIdx.x;
  if (i < n) p[i] = 0.0f;
}

// ---------------------------------------------------------------------------
// Weight prep: w [KW, COUT] fp32 -> wT [COUT, KPAD] f16 (K-padded with zeros).
// Makes each lane's WMMA B fragment two contiguous 16-byte global loads.
// ---------------------------------------------------------------------------
__global__ void weights_to_f16T(const float* __restrict__ w,
                                _Float16* __restrict__ wT, int KW, int KPAD,
                                int COUT) {
  const int idx = blockIdx.x * blockDim.x + threadIdx.x;
  if (idx >= COUT * KPAD) return;
  const int n = idx / KPAD, k = idx % KPAD;
  wT[idx] = (k < KW) ? (_Float16)w[k * COUT + n] : (_Float16)0.0f;
}

// ---------------------------------------------------------------------------
// Fused octree gather + WMMA GEMM (f16 in, f32 accum) + BN partial stats.
//   x     : [N, CIN]  fp32
//   neigh : [N, 27]   int32
//   wT    : [COUT, KPAD] f16 (pre-transposed, zero-padded)
//   y     : [N, COUT] fp32 (pre-BN conv output)
//   gsum/gsq : [COUT] global accumulators (atomic f32)
// Block = (ROWS/16) waves of 32; each wave owns a 16-row M tile.
// ---------------------------------------------------------------------------
template <int CIN, int COUT, int ROWS>
__global__ __launch_bounds__(ROWS * 2) void octree_conv_wmma(
    const float* __restrict__ x, const int* __restrict__ neigh,
    const _Float16* __restrict__ wT, float* __restrict__ y,
    float* __restrict__ gsum, float* __restrict__ gsq) {
  constexpr int KW   = KNB * CIN;
  constexpr int KPAD = ((KW + 31) / 32) * 32;
  constexpr int NT   = COUT / 16;
  constexpr int NTH  = ROWS * 2;

  __shared__ __align__(16) _Float16 As[ROWS * KPAD];
  __shared__ float s_sum[COUT];
  __shared__ float s_sq[COUT];

  const int tid = threadIdx.x;
  if (tid < COUT) { s_sum[tid] = 0.0f; s_sq[tid] = 0.0f; }

  const int row0 = blockIdx.x * ROWS;

  // stream-prefetch this block's neighbor-index slab
  __builtin_prefetch(&neigh[(row0 + (tid % ROWS)) * KNB], 0, 0);

  // cooperative gather: one (row, neighbor) pair per loop slot
  for (int p = tid; p < ROWS * KNB; p += NTH) {
    const int r = p / KNB, j = p % KNB;
    const int gi = neigh[(row0 + r) * KNB + j];
    const float* src = x + gi * CIN;
    _Float16* dst = &As[r * KPAD + j * CIN];
#pragma unroll
    for (int c = 0; c < CIN; ++c) dst[c] = (_Float16)src[c];
  }
  if (KPAD > KW) {
    for (int p = tid; p < ROWS * (KPAD - KW); p += NTH) {
      const int r = p / (KPAD - KW), k = KW + p % (KPAD - KW);
      As[r * KPAD + k] = (_Float16)0.0f;
    }
  }
  __syncthreads();

  const int wave = tid >> 5, lane = tid & 31;
  const int half = lane >> 4, m = lane & 15;
  const int rbase = wave * 16;

#pragma unroll
  for (int nt = 0; nt < NT; ++nt) {
    v8f acc = {};
    const int ncol = nt * 16 + m;  // this lane's output channel (C/D: N = lane%16)
    for (int cb = 0; cb < KPAD; cb += 32) {
      // A fragment: 16-bit A 16x32 layout — lane covers K in
      // [cb+half*8, +8) and [cb+16+half*8, +8): two contiguous b128 LDS loads
      const _Float16* ap = &As[(rbase + m) * KPAD + cb + half * 8];
      v8h a0 = *(const v8h*)ap;
      v8h a1 = *(const v8h*)(ap + 16);
      v16h a;
#pragma unroll
      for (int i = 0; i < 8; ++i) { a[i] = a0[i]; a[i + 8] = a1[i]; }
      // B fragment: lane (N = m) holds K in [cb+half*8, +8) and
      // [cb+16+half*8, +8) -> two contiguous b128 global loads from transposed f16 weights
      const _Float16* bp = &wT[ncol * KPAD + cb + half * 8];
      v8h b0 = *(const v8h*)bp;
      v8h b1 = *(const v8h*)(bp + 16);
      v16h bf;
#pragma unroll
      for (int i = 0; i < 8; ++i) { bf[i] = b0[i]; bf[i + 8] = b1[i]; }
      acc = __builtin_amdgcn_wmma_f32_16x16x32_f16(false, a, false, bf,
                                                   (short)0, acc, false, false);
    }
    // write y tile + per-channel BN partial sums (C/D: row = r + 8*half)
    float ls = 0.0f, lq = 0.0f;
#pragma unroll
    for (int r = 0; r < 8; ++r) {
      const float v = acc[r];
      ls += v; lq += v * v;
      const int grow = row0 + rbase + r + 8 * half;
      y[grow * COUT + ncol] = v;
    }
    atomicAdd(&s_sum[ncol], ls);
    atomicAdd(&s_sq[ncol], lq);
  }
  __syncthreads();
  if (tid < COUT) {
    atomicAdd(&gsum[tid], s_sum[tid]);
    atomicAdd(&gsq[tid], s_sq[tid]);
  }
}

// ---------------------------------------------------------------------------
// BN finalize: (sum, sumsq) -> (scale, shift)
// ---------------------------------------------------------------------------
__global__ void bn_finalize(const float* __restrict__ gsum,
                            const float* __restrict__ gsq,
                            const float* __restrict__ g,
                            const float* __restrict__ b,
                            float* __restrict__ scale,
                            float* __restrict__ shift, int C, float invN) {
  const int c = threadIdx.x;
  if (c >= C) return;
  const float mean = gsum[c] * invN;
  const float var  = gsq[c] * invN - mean * mean;
  const float s    = g[c] * rsqrtf(var + 1e-5f);
  scale[c] = s;
  shift[c] = b[c] - mean * s;
}

// ---------------------------------------------------------------------------
// BN + ReLU + octree max-pool (8 children -> 1 parent)
// ---------------------------------------------------------------------------
__global__ void bn_relu_pool(const float* __restrict__ y,
                             const float* __restrict__ scale,
                             const float* __restrict__ shift,
                             float* __restrict__ xout, int nPar, int C) {
  const int idx = blockIdx.x * blockDim.x + threadIdx.x;
  if (idx >= nPar * C) return;
  const int p = idx / C, c = idx % C;
  const float s = scale[c], t = shift[c];
  float mx = 0.0f;  // max(relu(v_k)) == max(0, max v_k)
#pragma unroll
  for (int k = 0; k < 8; ++k) {
    const float v = y[(p * 8 + k) * C + c] * s + t;
    mx = fmaxf(mx, v);
  }
  xout[idx] = mx;
}

// ---------------------------------------------------------------------------
// FullOctree2Voxel: x2 [16*64, 64] -> xv f16 [16, 4096], xv[b, c*64+k]=x2[b*64+k, c]
// ---------------------------------------------------------------------------
__global__ void voxelize_f16(const float* __restrict__ x2,
                             _Float16* __restrict__ xv, int total) {
  const int idx = blockIdx.x * blockDim.x + threadIdx.x;
  if (idx >= total) return;
  const int b = idx >> 12;       // /4096
  const int kk = idx & 4095;
  const int c = kk >> 6, k = kk & 63;
  xv[idx] = (_Float16)x2[((b << 6) + k) * 64 + c];
}

// ---------------------------------------------------------------------------
// Header: FC [16,4096]x[4096,128] via WMMA, +bias, batch BN (16), ReLU,
// then [16,128]x[128,40]+bias. One workgroup of 8 waves.
// wTfc: fc_w pre-transposed to [128, 4096] f16.
// ---------------------------------------------------------------------------
__global__ __launch_bounds__(256) void head_kernel(
    const _Float16* __restrict__ xv, const _Float16* __restrict__ wTfc,
    const float* __restrict__ fc_b, const float* __restrict__ bg,
    const float* __restrict__ bb, const float* __restrict__ ow,
    const float* __restrict__ ob, float* __restrict__ out) {
  __shared__ float yb[16 * 128];
  const int tid = threadIdx.x;
  const int wave = tid >> 5, lane = tid & 31;
  const int half = lane >> 4, m = lane & 15;
  const int ncol = wave * 16 + m;

  v8f acc = {};
  for (int cb = 0; cb < 4096; cb += 32) {
    const _Float16* ap = &xv[m * 4096 + cb + half * 8];
    v8h a0 = *(const v8h*)ap;
    v8h a1 = *(const v8h*)(ap + 16);
    v16h a;
#pragma unroll
    for (int i = 0; i < 8; ++i) { a[i] = a0[i]; a[i + 8] = a1[i]; }
    const _Float16* bp = &wTfc[ncol * 4096 + cb + half * 8];
    v8h b0 = *(const v8h*)bp;
    v8h b1 = *(const v8h*)(bp + 16);
    v16h bf;
#pragma unroll
    for (int i = 0; i < 8; ++i) { bf[i] = b0[i]; bf[i + 8] = b1[i]; }
    acc = __builtin_amdgcn_wmma_f32_16x16x32_f16(false, a, false, bf,
                                                 (short)0, acc, false, false);
  }
#pragma unroll
  for (int r = 0; r < 8; ++r)
    yb[(r + 8 * half) * 128 + ncol] = acc[r] + fc_b[ncol];
  __syncthreads();

  if (tid < 128) {
    float s = 0.0f, q = 0.0f;
#pragma unroll
    for (int r = 0; r < 16; ++r) {
      const float v = yb[r * 128 + tid];
      s += v; q += v * v;
    }
    const float mean = s * (1.0f / 16.0f);
    const float var  = q * (1.0f / 16.0f) - mean * mean;
    const float sc   = bg[tid] * rsqrtf(var + 1e-5f);
    const float sh   = bb[tid] - mean * sc;
#pragma unroll
    for (int r = 0; r < 16; ++r)
      yb[r * 128 + tid] = fmaxf(yb[r * 128 + tid] * sc + sh, 0.0f);
  }
  __syncthreads();

  for (int idx = tid; idx < 16 * 40; idx += 256) {
    const int b = idx / 40, o = idx % 40;
    float s = ob[o];
    for (int c = 0; c < 128; ++c) s += yb[b * 128 + c] * ow[c * 40 + o];
    out[idx] = s;
  }
}

// ---------------------------------------------------------------------------
// Host side
// ---------------------------------------------------------------------------
extern "C" void kernel_launch(void* const* d_in, const int* in_sizes, int n_in,
                              void* d_out, int out_size, void* d_ws,
                              size_t ws_size, hipStream_t stream) {
  (void)in_sizes; (void)n_in; (void)out_size; (void)ws_size;
  const float* feat   = (const float*)d_in[0];
  const int*   neigh5 = (const int*)d_in[1];
  const int*   neigh4 = (const int*)d_in[2];
  const int*   neigh3 = (const int*)d_in[3];
  const float* w5     = (const float*)d_in[4];
  const float* w4     = (const float*)d_in[5];
  const float* w3     = (const float*)d_in[6];
  const float* bn5_g  = (const float*)d_in[7];
  const float* bn5_b  = (const float*)d_in[8];
  const float* bn4_g  = (const float*)d_in[9];
  const float* bn4_b  = (const float*)d_in[10];
  const float* bn3_g  = (const float*)d_in[11];
  const float* bn3_b  = (const float*)d_in[12];
  const float* fc_w   = (const float*)d_in[13];
  const float* fc_b   = (const float*)d_in[14];
  const float* fcbn_g = (const float*)d_in[15];
  const float* fcbn_b = (const float*)d_in[16];
  const float* out_w  = (const float*)d_in[17];
  const float* out_b  = (const float*)d_in[18];
  float* out = (float*)d_out;

  // workspace carve-up (floats first, then 16B-aligned f16 region)
  float* f  = (float*)d_ws;
  float* y5 = f;                       // N5*16
  float* x4 = y5 + (size_t)N5 * 16;    // N4*16
  float* y4 = x4 + (size_t)N4 * 16;    // N4*32
  float* x3 = y4 + (size_t)N4 * 32;    // N3*32
  float* y3 = x3 + (size_t)N3 * 32;    // N3*64
  float* x2 = y3 + (size_t)N3 * 64;    // N2*64
  float* st = x2 + (size_t)N2 * 64;    // 3 stages x 4 arrays x 64 = 768 floats
  _Float16* xv   = (_Float16*)(st + 768);    // 16*4096      (16B aligned)
  _Float16* wT5  = xv + (size_t)16 * 4096;   // 16*96
  _Float16* wT4  = wT5 + (size_t)16 * 96;    // 32*448
  _Float16* wT3  = wT4 + (size_t)32 * 448;   // 64*864
  _Float16* wTfc = wT3 + (size_t)64 * 864;   // 128*4096

  float* s5sum = st + 0,   *s5sq = st + 64,  *s5sc = st + 128, *s5sh = st + 192;
  float* s4sum = st + 256, *s4sq = st + 320, *s4sc = st + 384, *s4sh = st + 448;
  float* s3sum = st + 512, *s3sq = st + 576, *s3sc = st + 640, *s3sh = st + 704;

  // zero BN stat accumulators (workspace is poisoned by the harness)
  zero_kernel<<<3, 256, 0, stream>>>(st, 768);

  // pre-transpose/convert all weight matrices to f16 [COUT, KPAD]
  weights_to_f16T<<<(16 * 96 + 255) / 256, 256, 0, stream>>>(w5, wT5, 81, 96, 16);
  weights_to_f16T<<<(32 * 448 + 255) / 256, 256, 0, stream>>>(w4, wT4, 432, 448, 32);
  weights_to_f16T<<<(64 * 864 + 255) / 256, 256, 0, stream>>>(w3, wT3, 864, 864, 64);
  weights_to_f16T<<<(128 * 4096 + 255) / 256, 256, 0, stream>>>(fc_w, wTfc, 4096, 4096, 128);

  // ---- depth 5: conv(3->16) + BN + ReLU + pool ----
  octree_conv_wmma<3, 16, 128><<<N5 / 128, 256, 0, stream>>>(
      feat, neigh5, wT5, y5, s5sum, s5sq);
  bn_finalize<<<1, 16, 0, stream>>>(s5sum, s5sq, bn5_g, bn5_b, s5sc, s5sh, 16,
                                    1.0f / (float)N5);
  bn_relu_pool<<<(N4 * 16 + 255) / 256, 256, 0, stream>>>(y5, s5sc, s5sh, x4,
                                                          N4, 16);

  // ---- depth 4: conv(16->32) + BN + ReLU + pool ----
  octree_conv_wmma<16, 32, 64><<<N4 / 64, 128, 0, stream>>>(
      x4, neigh4, wT4, y4, s4sum, s4sq);
  bn_finalize<<<1, 32, 0, stream>>>(s4sum, s4sq, bn4_g, bn4_b, s4sc, s4sh, 32,
                                    1.0f / (float)N4);
  bn_relu_pool<<<(N3 * 32 + 255) / 256, 256, 0, stream>>>(y4, s4sc, s4sh, x3,
                                                          N3, 32);

  // ---- depth 3: conv(32->64) + BN + ReLU + pool ----
  octree_conv_wmma<32, 64, 32><<<N3 / 32, 64, 0, stream>>>(
      x3, neigh3, wT3, y3, s3sum, s3sq);
  bn_finalize<<<1, 64, 0, stream>>>(s3sum, s3sq, bn3_g, bn3_b, s3sc, s3sh, 64,
                                    1.0f / (float)N3);
  bn_relu_pool<<<(N2 * 64 + 255) / 256, 256, 0, stream>>>(y3, s3sc, s3sh, x2,
                                                          N2, 64);

  // ---- voxelize + header ----
  voxelize_f16<<<(16 * 4096 + 255) / 256, 256, 0, stream>>>(x2, xv, 16 * 4096);
  head_kernel<<<1, 256, 0, stream>>>(xv, wTfc, fc_b, fcbn_g, fcbn_b, out_w,
                                     out_b, out);
}
